// SpatialAttentionFusionDeforbaleMask_26199300505714
// MI455X (gfx1250) — matrix operand order repaired and checked by reading, not compiled, not run on hardware
//
#include <hip/hip_runtime.h>
#include <cmath>

typedef __attribute__((ext_vector_type(16))) _Float16 v16h;
typedef __attribute__((ext_vector_type(8)))  float    v8f;

// ---------------------------------------------------------------------------
// Pack fp32 conv weights (CoutSrc, C, 3,3) -> f16 wp[o][t*C + c], t = ky*3+kx,
// zero-padding output rows up to CoutPad.
// ---------------------------------------------------------------------------
__global__ __launch_bounds__(256)
void pack_weights_kernel(const float* __restrict__ w, _Float16* __restrict__ wp,
                         int CoutSrc, int CoutPad, int C) {
  int idx = blockIdx.x * blockDim.x + threadIdx.x;
  int total = CoutPad * C * 9;
  if (idx >= total) return;
  int o = idx / (C * 9);
  int r = idx % (C * 9);
  int t = r / C;
  int c = r % C;
  float v = (o < CoutSrc) ? w[((size_t)o * C + c) * 9 + t] : 0.f;
  wp[(size_t)o * (9 * C) + (size_t)t * C + c] = (_Float16)v;
}

// ---------------------------------------------------------------------------
// WMMA GEMM over K = 9*C (k = tap*C + c).
//   DEFORM=true : taps at clamped bilinear positions from `off` (deform conv)
//   DEFORM=false: taps at integer grid positions, zero-pad boundary (3x3 conv)
// Block = OTILES*64 threads = OTILES*2 waves: 2 pixel-subtiles x OTILES
// o-subtiles -> tile = 32 pixels x OTILES*16 channels.
// Double-buffered LDS sample tile; one barrier per K-chunk.
// ---------------------------------------------------------------------------
template <int OTILES, bool DEFORM>
__global__ __launch_bounds__(OTILES * 64)
void deform_wmma_kernel(const float* __restrict__ x, const float* __restrict__ off,
                        const _Float16* __restrict__ wp, const float* __restrict__ bias,
                        float* __restrict__ z, int C, int H, int W,
                        int CoutStride, int CoutStore) {
  const int HW = H * W;
  const int tilesPerB = HW >> 5;
  const int b    = blockIdx.x / tilesPerB;
  const int tile = blockIdx.x % tilesPerB;
  const int pix0 = tile << 5;              // 32 pixels per block
  const int tid  = threadIdx.x;
  const int lane = tid & 31;
  const int wave = tid >> 5;
  const int nt   = wave & 1;               // pixel subtile (0/1)
  const int ot   = wave >> 1;              // o subtile
  const int och0 = blockIdx.y * (OTILES * 16) + ot * 16;

  __shared__ _Float16 sB[2][32][40];       // [buf][pixel][k] (+8 pad)
  __shared__ float    s_w4[32][4];
  __shared__ int      s_i4[32][4];

  v8f acc = {};

  constexpr int TPP  = OTILES * 2;         // gather threads per pixel
  constexpr int KPER = 32 / TPP;           // k-values per gather thread
  const int gn  = tid / TPP;               // pixel 0..31
  const int gk0 = (tid % TPP) * KPER;
  const float* xb = x + (size_t)b * C * HW;

  int q = 0;                                // global chunk counter (parity)
  for (int t = 0; t < 9; ++t) {
    if (tid < 32) {
      int p  = pix0 + tid;
      int yi = p / W, xi = p % W;
      if (DEFORM) {
        float oy = off[((size_t)b * 18 + t) * HW + p];
        float ox = off[((size_t)b * 18 + 9 + t) * HW + p];
        float py = (float)yi + (float)(t / 3 - 1) + oy;
        float px = (float)xi + (float)(t % 3 - 1) + ox;
        py = fminf(fmaxf(py, 0.f), (float)(H - 1));
        px = fminf(fmaxf(px, 0.f), (float)(W - 1));
        float y0f = floorf(py), x0f = floorf(px);
        float wy = py - y0f, wx = px - x0f;
        int y0 = max(0, min((int)y0f, H - 1));
        int x0 = max(0, min((int)x0f, W - 1));
        int y1 = min(y0 + 1, H - 1), x1 = min(x0 + 1, W - 1);
        s_i4[tid][0] = y0 * W + x0;
        s_i4[tid][1] = y0 * W + x1;
        s_i4[tid][2] = y1 * W + x0;
        s_i4[tid][3] = y1 * W + x1;
        s_w4[tid][0] = (1.f - wy) * (1.f - wx);
        s_w4[tid][1] = (1.f - wy) * wx;
        s_w4[tid][2] = wy * (1.f - wx);
        s_w4[tid][3] = wy * wx;
      } else {
        int ty = yi + (t / 3 - 1), tx = xi + (t % 3 - 1);
        bool in = (ty >= 0) & (ty < H) & (tx >= 0) & (tx < W);
        int p0 = in ? (ty * W + tx) : 0;
        s_i4[tid][0] = p0; s_i4[tid][1] = p0;
        s_i4[tid][2] = p0; s_i4[tid][3] = p0;
        s_w4[tid][0] = in ? 1.f : 0.f;
        s_w4[tid][1] = 0.f; s_w4[tid][2] = 0.f; s_w4[tid][3] = 0.f;
      }
    }
    __syncthreads();
    const int   i0 = s_i4[gn][0], i1 = s_i4[gn][1], i2 = s_i4[gn][2], i3 = s_i4[gn][3];
    const float w0 = s_w4[gn][0], w1 = s_w4[gn][1], w2 = s_w4[gn][2], w3 = s_w4[gn][3];

    for (int cb = 0; cb < C; cb += 32, ++q) {
      const int par = q & 1;
      // cooperative bilinear gather: 32 K x 32 pixels -> LDS (f16)
      const float* xc = xb + (size_t)(cb + gk0) * HW;
#pragma unroll
      for (int e = 0; e < KPER; ++e) {
        float v = w0 * xc[i0] + w1 * xc[i1] + w2 * xc[i2] + w3 * xc[i3];
        sB[par][gn][gk0 + e] = (_Float16)v;
        xc += HW;
      }
      __syncthreads();

      // A fragment (weights): contiguous K = h*16 + e order; two contiguous 16B chunks.
      const int h = lane >> 4;
      const int m = och0 + (lane & 15);
      const _Float16* ap = wp + (size_t)m * (9 * C) + (size_t)t * C + cb;
      v16h a, bf;
#pragma unroll
      for (int e = 0; e < 8; ++e) {
        a[e]     = ap[h * 16 + e];      // K = h*16 + e
        a[8 + e] = ap[h * 16 + 8 + e];  // K = h*16 + 8 + e
      }
      // B fragment (samples): lane = column; lanes>=16 hold K=16..31.
      const _Float16* bp = &sB[par][nt * 16 + (lane & 15)][h * 16];
#pragma unroll
      for (int e = 0; e < 16; ++e) bf[e] = bp[e];

      acc = __builtin_amdgcn_wmma_f32_16x16x32_f16(
          /*neg_a=*/false, a, /*neg_b=*/false, bf,
          /*c_mod=*/(short)0, acc, /*reuse_a=*/false, /*reuse_b=*/false);
    }
  }

  // C/D layout: lane = column(pixel); VGPR v -> row v + 8*(lane>=16)
  const int p  = pix0 + nt * 16 + (lane & 15);
  const int mb = (lane >> 4) * 8;
#pragma unroll
  for (int v = 0; v < 8; ++v) {
    int o = och0 + mb + v;
    if (o < CoutStore)
      z[((size_t)b * CoutStride + o) * HW + p] = acc[v] + bias[o];
  }
}

// ---------------------------------------------------------------------------
// GroupNorm (32 groups), in-place. One block per (batch, group).
// ---------------------------------------------------------------------------
__global__ __launch_bounds__(256)
void groupnorm_kernel(float* __restrict__ y, const float* __restrict__ gamma,
                      const float* __restrict__ beta, int Cout, int HW) {
  int bg = blockIdx.x;
  int b = bg >> 5, g = bg & 31;
  int cpg = Cout / 32;
  size_t base = ((size_t)b * Cout + (size_t)g * cpg) * HW;
  int n = cpg * HW;
  float s = 0.f, s2 = 0.f;
  for (int idx = threadIdx.x; idx < n; idx += blockDim.x) {
    float v = y[base + idx];
    s += v; s2 += v * v;
  }
  __shared__ float rs[256], rq[256];
  rs[threadIdx.x] = s; rq[threadIdx.x] = s2;
  __syncthreads();
  for (int st = 128; st > 0; st >>= 1) {
    if ((int)threadIdx.x < st) {
      rs[threadIdx.x] += rs[threadIdx.x + st];
      rq[threadIdx.x] += rq[threadIdx.x + st];
    }
    __syncthreads();
  }
  __shared__ float mu_s, rstd_s;
  if (threadIdx.x == 0) {
    float mu = rs[0] / (float)n;
    float var = rq[0] / (float)n - mu * mu;
    mu_s = mu;
    rstd_s = rsqrtf(var + 1e-5f);
  }
  __syncthreads();
  float mu = mu_s, rstd = rstd_s;
  for (int idx = threadIdx.x; idx < n; idx += blockDim.x) {
    int c = g * cpg + idx / HW;
    float v = y[base + idx];
    y[base + idx] = (v - mu) * rstd * gamma[c] + beta[c];
  }
}

// ---------------------------------------------------------------------------
// Mask: sigmoid(conv3x3(y1, w2)+b2), 1 output channel. msk[b*HW+p]
// ---------------------------------------------------------------------------
__global__ __launch_bounds__(256)
void mask_kernel(const float* __restrict__ y1, const float* __restrict__ w2,
                 const float* __restrict__ b2, float* __restrict__ msk,
                 int Cout, int H, int W, int B) {
  int HW = H * W;
  int idx = blockIdx.x * blockDim.x + threadIdx.x;
  if (idx >= B * HW) return;
  int b = idx / HW, p = idx % HW;
  int y = p / W, x = p % W;
  float acc = b2[0];
  for (int c = 0; c < Cout; ++c) {
    const float* yc = y1 + ((size_t)b * Cout + c) * HW;
    const float* wc = w2 + (size_t)c * 9;
#pragma unroll
    for (int ky = 0; ky < 3; ++ky) {
      int yy = y + ky - 1;
      if (yy < 0 || yy >= H) continue;
#pragma unroll
      for (int kx = 0; kx < 3; ++kx) {
        int xv = x + kx - 1;
        if (xv < 0 || xv >= W) continue;
        acc += yc[yy * W + xv] * wc[ky * 3 + kx];
      }
    }
  }
  msk[idx] = 1.f / (1.f + expf(-acc));
}

// ---------------------------------------------------------------------------
// Fused: out = sum_j wj * resize(y1_j * msk_j)  (align-corners bilinear)
// ---------------------------------------------------------------------------
__device__ __forceinline__ float sample_branch(const float* __restrict__ y,
                                               const float* __restrict__ m,
                                               int H, int W, int HT, int WT,
                                               int b, int o, int yt, int xt, int Cout) {
  int HW = H * W;
  const float* yc = y + ((size_t)b * Cout + o) * HW;
  const float* mb = m + (size_t)b * HW;
  if (H == HT && W == WT) {
    int p = yt * W + xt;
    return yc[p] * mb[p];
  }
  float sy = (float)(H - 1) / (float)(HT - 1);
  float sx = (float)(W - 1) / (float)(WT - 1);
  float ys = yt * sy, xsv = xt * sx;
  int y0 = (int)floorf(ys); int y1 = min(y0 + 1, H - 1);
  float wy = ys - (float)y0;
  int x0 = (int)floorf(xsv); int x1 = min(x0 + 1, W - 1);
  float wx = xsv - (float)x0;
  int p00 = y0 * W + x0, p01 = y0 * W + x1, p10 = y1 * W + x0, p11 = y1 * W + x1;
  float g00 = yc[p00] * mb[p00], g01 = yc[p01] * mb[p01];
  float g10 = yc[p10] * mb[p10], g11 = yc[p11] * mb[p11];
  return g00 * (1.f - wy) * (1.f - wx) + g01 * (1.f - wy) * wx +
         g10 * wy * (1.f - wx) + g11 * wy * wx;
}

__global__ __launch_bounds__(256)
void fuse_kernel(const float* __restrict__ ya, const float* __restrict__ ma, int Ha, int Wa,
                 const float* __restrict__ yb, const float* __restrict__ mb, int Hb, int Wb,
                 const float* __restrict__ yc, const float* __restrict__ mc, int Hc, int Wc,
                 float w0, float w1, float w2,
                 float* __restrict__ out, int Cout, int HT, int WT, int B) {
  int idx = blockIdx.x * blockDim.x + threadIdx.x;
  int total = B * Cout * HT * WT;
  if (idx >= total) return;
  int xt = idx % WT;
  int yt = (idx / WT) % HT;
  int o  = (idx / (WT * HT)) % Cout;
  int b  = idx / (WT * HT * Cout);
  float acc = w0 * sample_branch(ya, ma, Ha, Wa, HT, WT, b, o, yt, xt, Cout)
            + w1 * sample_branch(yb, mb, Hb, Wb, HT, WT, b, o, yt, xt, Cout)
            + w2 * sample_branch(yc, mc, Hc, Wc, HT, WT, b, o, yt, xt, Cout);
  out[idx] = acc;
}

// ---------------------------------------------------------------------------
// Host driver
// ---------------------------------------------------------------------------
extern "C" void kernel_launch(void* const* d_in, const int* in_sizes, int n_in,
                              void* d_out, int out_size, void* d_ws, size_t ws_size,
                              hipStream_t stream) {
  (void)in_sizes; (void)n_in; (void)out_size; (void)ws_size;
  const int CHv[3] = {128, 256, 512};
  const int HHv[3] = {80, 40, 20};
  const int WWv[3] = {80, 40, 20};
  const int B = 4;
  const float WMATf[3][3] = {{0.50f, 0.25f, 0.25f},
                             {0.25f, 0.50f, 0.25f},
                             {0.50f, 0.25f, 0.25f}};

  const float* xs[3] = {(const float*)d_in[0], (const float*)d_in[1], (const float*)d_in[2]};
  auto P = [&](int bidx, int leaf) -> const float* {
    return (const float*)d_in[3 + bidx * 8 + leaf];
  };

  // Workspace partition
  size_t pos = 0;
  auto take = [&](size_t bytes) -> void* {
    void* p = (char*)d_ws + pos;
    pos += (bytes + 255) & ~(size_t)255;
    return p;
  };
  _Float16* wp[9];   // main deform weights, [Cout][9*C]
  _Float16* wpo[9];  // offset-conv weights, [32 (18 padded)][9*C]
  for (int bi = 0; bi < 9; ++bi) {
    int Co = CHv[bi / 3], Ci = CHv[bi % 3];
    wp[bi]  = (_Float16*)take((size_t)Co * 9 * Ci * sizeof(_Float16));
    wpo[bi] = (_Float16*)take((size_t)32 * 9 * Ci * sizeof(_Float16));
  }
  float* offbuf[3];
  float* y1buf[3];
  float* mskbuf[3];
  for (int j = 0; j < 3; ++j) {
    size_t HW = (size_t)HHv[j] * WWv[j];
    offbuf[j] = (float*)take((size_t)B * 18 * HW * sizeof(float));
    y1buf[j]  = (float*)take((size_t)B * 512 * HW * sizeof(float));  // max Cout
    mskbuf[j] = (float*)take((size_t)B * HW * sizeof(float));
  }

  // Pack all branch weights to f16 (k = tap*C + c ordering)
  for (int bi = 0; bi < 9; ++bi) {
    int Co = CHv[bi / 3], Ci = CHv[bi % 3];
    int tot1 = Co * Ci * 9;
    pack_weights_kernel<<<(tot1 + 255) / 256, 256, 0, stream>>>(
        P(bi, 2), wp[bi], Co, Co, Ci);
    int tot2 = 32 * Ci * 9;
    pack_weights_kernel<<<(tot2 + 255) / 256, 256, 0, stream>>>(
        P(bi, 0), wpo[bi], 18, 32, Ci);
  }

  float* outp[3];
  outp[0] = (float*)d_out;
  outp[1] = outp[0] + (size_t)B * 128 * 80 * 80;
  outp[2] = outp[1] + (size_t)B * 256 * 40 * 40;

  for (int i = 0; i < 3; ++i) {
    int Cout = CHv[i];
    for (int j = 0; j < 3; ++j) {
      int bidx = i * 3 + j;
      int C = CHv[j], H = HHv[j], W = WWv[j], HW = H * W;

      // offset conv (18ch zero-pad 3x3) on WMMA: 32 pixels x 32 channels/block
      dim3 g1(B * (HW / 32), 1);
      deform_wmma_kernel<2, false><<<g1, 128, 0, stream>>>(
          xs[j], nullptr, wpo[bidx], P(bidx, 1), offbuf[j], C, H, W, 18, 18);

      // deformable conv GEMM: 32 pixels x 64 channels/block
      dim3 g2(B * (HW / 32), Cout / 64);
      deform_wmma_kernel<4, true><<<g2, 256, 0, stream>>>(
          xs[j], offbuf[j], wp[bidx], P(bidx, 3), y1buf[j], C, H, W, Cout, Cout);

      groupnorm_kernel<<<B * 32, 256, 0, stream>>>(
          y1buf[j], P(bidx, 4), P(bidx, 5), Cout, HW);

      int tot2 = B * HW;
      mask_kernel<<<(tot2 + 255) / 256, 256, 0, stream>>>(
          y1buf[j], P(bidx, 6), P(bidx, 7), mskbuf[j], Cout, H, W, B);
    }
    int HT = HHv[i], WT = WWv[i];
    int tot3 = B * Cout * HT * WT;
    fuse_kernel<<<(tot3 + 255) / 256, 256, 0, stream>>>(
        y1buf[0], mskbuf[0], HHv[0], WWv[0],
        y1buf[1], mskbuf[1], HHv[1], WWv[1],
        y1buf[2], mskbuf[2], HHv[2], WWv[2],
        WMATf[i][0], WMATf[i][1], WMATf[i][2],
        outp[i], Cout, HT, WT, B);
  }
}
